// VectorQuantizer_1151051236002
// MI455X (gfx1250) — compile-verified, matrix operand order checked by reading, not run on hardware
//
#include <hip/hip_runtime.h>
#include <hip/hip_fp16.h>
#include <math.h>

// ---------------------------------------------------------------------------
// VQ + Sinkhorn for MI455X (gfx1250, wave32, WMMA).
// Sinkhorn factorized as Q = diag(r) * A * diag(c); A = exp(-(t - tmin_i)*alpha_i)
// with t_ij = ||e_j||^2 - 2 x_i.e_j, alpha_i = 1/(std_i * eps).  A tiles are
// recomputed on the fly with v_wmma_f32_16x16x32_f16 (emb staged in 288KB LDS),
// so no 134-256MB Q buffer is ever materialized; per-pass traffic is ~5MB
// (x_f16, L2-resident) and the loop is exp/WMMA bound, far under the 23.3TB/s
// HBM roofline of the naive 100x-streamed-Q formulation.
// ---------------------------------------------------------------------------

#define TOK   32768
#define NE    2048
#define ED    64
#define SKITERS 100
static constexpr float SK_EPS = 0.003f;

typedef __attribute__((ext_vector_type(16))) _Float16 v16h;
typedef __attribute__((ext_vector_type(8)))  _Float16 v8h;
typedef __attribute__((ext_vector_type(8)))  float    v8f;

#define ROWS_PER_WG 128                  // 8 waves x 16-row WMMA tile
#define NSTRIPE     (TOK / ROWS_PER_WG)  // 256 workgroups
#define NTILES      (NE / 16)            // 128 column tiles
#define LSTRIDE     72                   // halves per code row in LDS (bank padding)

#define SM_EMB_BYTES (NE * LSTRIDE * 2)        // 294912
#define SM_ESQ_OFF   SM_EMB_BYTES              // float[2048]
#define SM_AUX_OFF   (SM_ESQ_OFF + NE * 4)     // float[2048]  (vws / c / log c)
#define SM_JBUF_OFF  (SM_AUX_OFF + NE * 4)     // int[128]
#define SM_WSUM_OFF  (SM_JBUF_OFF + 512)       // float[8]
#define SM_BYTES     (SM_WSUM_OFF + 64)        // 311872 <= 320KB WGP LDS

// ------------------------------- helpers -----------------------------------

__device__ __forceinline__ void stage_emb_lds(char* smem,
                                              const _Float16* __restrict__ eh,
                                              const float* __restrict__ esq,
                                              int tid, int nthr) {
  unsigned* le = (unsigned*)smem;
  const unsigned* src = (const unsigned*)eh;
  for (int d = tid; d < NE * (ED / 2); d += nthr) {
    int code = d >> 5, kp = d & 31;
    le[code * (LSTRIDE / 2) + kp] = src[d];
  }
  float* lesq = (float*)(smem + SM_ESQ_OFF);
  for (int j = tid; j < NE; j += nthr) lesq[j] = esq[j];
}

// A-operand: x tile rows row_base..+15, K-step kstep (0: K 0..31, 1: K 32..63).
// ISA 16-bit A layout: lanes 0-15 -> K {0..7,16..23}, lanes 16-31 -> K {8..15,24..31}.
__device__ __forceinline__ v16h load_a_xtile(const _Float16* __restrict__ xh,
                                             int row_base, int kstep, int lane) {
  int lm = lane & 15, half = lane >> 4;
  const _Float16* p = xh + (size_t)(row_base + lm) * ED + half * 8 + kstep * 32;
  v8h lo = *(const v8h*)(p);
  v8h hi = *(const v8h*)(p + 16);
  return __builtin_shufflevector(lo, hi, 0,1,2,3,4,5,6,7,8,9,10,11,12,13,14,15);
}

// B-operand: emb^T tile (K x N), N = code col_base+lm; emb rows are the K axis
// so chunks are contiguous in LDS -> ds_load_b128.
__device__ __forceinline__ v16h load_b_etile(const _Float16* lemb,
                                             int col_base, int kstep, int lane) {
  int lm = lane & 15, half = lane >> 4;
  const _Float16* p = lemb + (col_base + lm) * LSTRIDE + half * 8 + kstep * 32;
  v8h lo = *(const v8h*)(p);
  v8h hi = *(const v8h*)(p + 16);
  return __builtin_shufflevector(lo, hi, 0,1,2,3,4,5,6,7,8,9,10,11,12,13,14,15);
}

// s(16x16) = x_tile(16x64) @ emb_tile(16x64)^T via two chained K=32 WMMAs.
__device__ __forceinline__ v8f mma_tile(v16h a0, v16h a1, const _Float16* lemb,
                                        int col_base, int lane) {
  v8f acc = {};
  v16h b0 = load_b_etile(lemb, col_base, 0, lane);
  acc = __builtin_amdgcn_wmma_f32_16x16x32_f16(false, a0, false, b0,
                                               (short)0, acc, false, false);
  v16h b1 = load_b_etile(lemb, col_base, 1, lane);
  acc = __builtin_amdgcn_wmma_f32_16x16x32_f16(false, a1, false, b1,
                                               (short)0, acc, false, false);
  return acc;
}

// ------------------------------- prep --------------------------------------

__global__ void prep_x_kernel(const float* __restrict__ x, _Float16* __restrict__ xh,
                              float* __restrict__ u, float* __restrict__ loss_slot) {
  int i = blockIdx.x * blockDim.x + threadIdx.x;     // 0 .. TOK*ED-1
  xh[i] = (_Float16)x[i];
  if (i < TOK) u[i] = 1.0f / (float)TOK;             // => r = 1 on iteration 0
  if (i == 0) *loss_slot = 0.0f;
}

__global__ void prep_emb_kernel(const float* __restrict__ emb, _Float16* __restrict__ eh,
                                float* __restrict__ esq) {
  int tid = threadIdx.x, lane = tid & 31, wave = tid >> 5;
  int code = blockIdx.x * 8 + wave;
  float a = emb[code * ED + lane];
  float b = emb[code * ED + lane + 32];
  eh[code * ED + lane]      = (_Float16)a;
  eh[code * ED + lane + 32] = (_Float16)b;
  float ss = a * a + b * b;
  #pragma unroll
  for (int m = 1; m < 32; m <<= 1) ss += __shfl_xor(ss, m, 32);
  if (lane == 0) esq[code] = ss;
}

// --------------------- per-row std (ddof=1) + min ---------------------------

__global__ void __launch_bounds__(256)
rowstat_kernel(const _Float16* __restrict__ xh, const _Float16* __restrict__ eh,
               const float* __restrict__ esq, float* __restrict__ alpha,
               float* __restrict__ tmin) {
  extern __shared__ char smem[];
  stage_emb_lds(smem, eh, esq, threadIdx.x, blockDim.x);
  __syncthreads();
  const _Float16* lemb = (const _Float16*)smem;
  const float* lesq = (const float*)(smem + SM_ESQ_OFF);

  int tid = threadIdx.x, lane = tid & 31, wave = tid >> 5;
  int lm = lane & 15, half = lane >> 4;
  int row_base = blockIdx.x * ROWS_PER_WG + wave * 16;
  v16h a0 = load_a_xtile(xh, row_base, 0, lane);
  v16h a1 = load_a_xtile(xh, row_base, 1, lane);

  float s1[8], s2[8], mn[8];
  #pragma unroll
  for (int g = 0; g < 8; ++g) { s1[g] = 0.f; s2[g] = 0.f; mn[g] = 3.402823466e38f; }

  for (int it = 0; it < NTILES; ++it) {
    int ct = (it + (wave << 4)) & (NTILES - 1);   // stagger waves -> disjoint LDS tiles
    v8f s = mma_tile(a0, a1, lemb, ct * 16, lane);
    float ej = lesq[ct * 16 + lm];
    #pragma unroll
    for (int g = 0; g < 8; ++g) {
      float t = ej - 2.0f * s[g];
      s1[g] += t; s2[g] += t * t; mn[g] = fminf(mn[g], t);
    }
  }
  #pragma unroll
  for (int m = 1; m < 16; m <<= 1) {
    #pragma unroll
    for (int g = 0; g < 8; ++g) {
      s1[g] += __shfl_xor(s1[g], m, 32);
      s2[g] += __shfl_xor(s2[g], m, 32);
      mn[g] = fminf(mn[g], __shfl_xor(mn[g], m, 32));
    }
  }
  if (lm == 0) {
    #pragma unroll
    for (int g = 0; g < 8; ++g) {
      int row = row_base + half * 8 + g;
      const float K = (float)NE;
      float var = (s2[g] - s1[g] * s1[g] / K) / (K - 1.0f);
      float sd = fmaxf(sqrtf(fmaxf(var, 0.0f)), 1e-6f);
      alpha[row] = 1.0f / (sd * SK_EPS);
      tmin[row] = mn[g];
    }
  }
}

// ------------------- Sinkhorn half-step 1: v_part = A^T r -------------------

__global__ void __launch_bounds__(256)
colsum_kernel(const _Float16* __restrict__ xh, const _Float16* __restrict__ eh,
              const float* __restrict__ esq, const float* __restrict__ alpha,
              const float* __restrict__ tmin, const float* __restrict__ u,
              float* __restrict__ v_part) {
  extern __shared__ char smem[];
  float* vws = (float*)(smem + SM_AUX_OFF);
  stage_emb_lds(smem, eh, esq, threadIdx.x, blockDim.x);
  for (int j = threadIdx.x; j < NE; j += blockDim.x) vws[j] = 0.0f;
  __syncthreads();
  const _Float16* lemb = (const _Float16*)smem;
  const float* lesq = (const float*)(smem + SM_ESQ_OFF);

  int tid = threadIdx.x, lane = tid & 31, wave = tid >> 5;
  int lm = lane & 15, half = lane >> 4;
  int row_base = blockIdx.x * ROWS_PER_WG + wave * 16;
  v16h a0 = load_a_xtile(xh, row_base, 0, lane);
  v16h a1 = load_a_xtile(xh, row_base, 1, lane);

  float rr[8], al[8], tm[8];
  #pragma unroll
  for (int g = 0; g < 8; ++g) {
    int row = row_base + half * 8 + g;
    rr[g] = 1.0f / ((float)TOK * u[row]);   // r_i = 1/(B * u_i)
    al[g] = alpha[row];
    tm[g] = tmin[row];
  }

  for (int it = 0; it < NTILES; ++it) {
    int ct = (it + (wave << 4)) & (NTILES - 1);   // waves hit disjoint vws ranges
    v8f s = mma_tile(a0, a1, lemb, ct * 16, lane);
    float ej = lesq[ct * 16 + lm];
    float part = 0.0f;
    #pragma unroll
    for (int g = 0; g < 8; ++g) {
      float t = ej - 2.0f * s[g];
      float A = __expf((tm[g] - t) * al[g]);   // in (0,1]
      part += A * rr[g];
    }
    part += __shfl_xor(part, 16, 32);          // combine halves: one atomic / column
    if (half == 0) atomicAdd(&vws[ct * 16 + lm], part);   // ds_add_f32
  }
  __syncthreads();
  float* vp = v_part + (size_t)blockIdx.x * NE;
  for (int j = threadIdx.x; j < NE; j += blockDim.x) vp[j] = vws[j];
}

// -------------------- reduce partials -> c = 1/(K v), log c -----------------

__global__ void reduce_c_kernel(const float* __restrict__ v_part,
                                float* __restrict__ c, float* __restrict__ lc) {
  int j = blockIdx.x * blockDim.x + threadIdx.x;   // 0..NE-1
  float s = 0.0f;
  for (int p = 0; p < NSTRIPE; ++p) s += v_part[(size_t)p * NE + j];
  float denom = fmaxf((float)NE * s, 1e-30f);
  c[j]  = 1.0f / denom;
  lc[j] = -__logf(denom);
}

// ------------------- Sinkhorn half-step 2: u = A c --------------------------

__global__ void __launch_bounds__(256)
rowsum_kernel(const _Float16* __restrict__ xh, const _Float16* __restrict__ eh,
              const float* __restrict__ esq, const float* __restrict__ alpha,
              const float* __restrict__ tmin, const float* __restrict__ c_in,
              float* __restrict__ u_out) {
  extern __shared__ char smem[];
  float* lcv = (float*)(smem + SM_AUX_OFF);
  stage_emb_lds(smem, eh, esq, threadIdx.x, blockDim.x);
  for (int j = threadIdx.x; j < NE; j += blockDim.x) lcv[j] = c_in[j];
  __syncthreads();
  const _Float16* lemb = (const _Float16*)smem;
  const float* lesq = (const float*)(smem + SM_ESQ_OFF);

  int tid = threadIdx.x, lane = tid & 31, wave = tid >> 5;
  int lm = lane & 15, half = lane >> 4;
  int row_base = blockIdx.x * ROWS_PER_WG + wave * 16;
  v16h a0 = load_a_xtile(xh, row_base, 0, lane);
  v16h a1 = load_a_xtile(xh, row_base, 1, lane);

  float al[8], tm[8], acc[8];
  #pragma unroll
  for (int g = 0; g < 8; ++g) {
    int row = row_base + half * 8 + g;
    al[g] = alpha[row]; tm[g] = tmin[row]; acc[g] = 0.0f;
  }

  for (int it = 0; it < NTILES; ++it) {
    int ct = (it + (wave << 4)) & (NTILES - 1);
    v8f s = mma_tile(a0, a1, lemb, ct * 16, lane);
    float ej = lesq[ct * 16 + lm];
    float cc = lcv[ct * 16 + lm];
    #pragma unroll
    for (int g = 0; g < 8; ++g) {
      float t = ej - 2.0f * s[g];
      acc[g] += __expf((tm[g] - t) * al[g]) * cc;
    }
  }
  #pragma unroll
  for (int m = 1; m < 16; m <<= 1) {
    #pragma unroll
    for (int g = 0; g < 8; ++g) acc[g] += __shfl_xor(acc[g], m, 32);
  }
  if (lm == 0) {
    #pragma unroll
    for (int g = 0; g < 8; ++g) u_out[row_base + half * 8 + g] = acc[g];
  }
}

// -------- final: argmax_j (log c_j - (t-tmin)*alpha), gather, loss ----------

__global__ void __launch_bounds__(256)
argmax_out_kernel(const float* __restrict__ x, const float* __restrict__ emb,
                  const _Float16* __restrict__ xh, const _Float16* __restrict__ eh,
                  const float* __restrict__ esq, const float* __restrict__ alpha,
                  const float* __restrict__ tmin, const float* __restrict__ lc_in,
                  float* __restrict__ out_xq, float* __restrict__ out_loss,
                  float* __restrict__ out_idx) {
  extern __shared__ char smem[];
  float* llc  = (float*)(smem + SM_AUX_OFF);
  int*   jbuf = (int*)(smem + SM_JBUF_OFF);
  float* wsum = (float*)(smem + SM_WSUM_OFF);
  stage_emb_lds(smem, eh, esq, threadIdx.x, blockDim.x);
  for (int j = threadIdx.x; j < NE; j += blockDim.x) llc[j] = lc_in[j];
  __syncthreads();
  const _Float16* lemb = (const _Float16*)smem;
  const float* lesq = (const float*)(smem + SM_ESQ_OFF);

  int tid = threadIdx.x, lane = tid & 31, wave = tid >> 5;
  int lm = lane & 15, half = lane >> 4;
  int row_base = blockIdx.x * ROWS_PER_WG + wave * 16;
  v16h a0 = load_a_xtile(xh, row_base, 0, lane);
  v16h a1 = load_a_xtile(xh, row_base, 1, lane);

  float al[8], tm[8], bv[8]; int bi[8];
  #pragma unroll
  for (int g = 0; g < 8; ++g) {
    int row = row_base + half * 8 + g;
    al[g] = alpha[row]; tm[g] = tmin[row];
    bv[g] = -3.402823466e38f; bi[g] = 0;
  }

  for (int it = 0; it < NTILES; ++it) {
    int ct = (it + (wave << 4)) & (NTILES - 1);
    v8f s = mma_tile(a0, a1, lemb, ct * 16, lane);
    int j = ct * 16 + lm;
    float ej = lesq[j];
    float lcj = llc[j];
    #pragma unroll
    for (int g = 0; g < 8; ++g) {
      float score = lcj - (ej - 2.0f * s[g] - tm[g]) * al[g];   // = log(c_j * A_ij)
      if (score > bv[g] || (score == bv[g] && j < bi[g])) { bv[g] = score; bi[g] = j; }
    }
  }
  #pragma unroll
  for (int m = 1; m < 16; m <<= 1) {
    #pragma unroll
    for (int g = 0; g < 8; ++g) {
      float ov = __shfl_xor(bv[g], m, 32);
      int   oi = __shfl_xor(bi[g], m, 32);
      if (ov > bv[g] || (ov == bv[g] && oi < bi[g])) { bv[g] = ov; bi[g] = oi; }
    }
  }
  if (lm == 0) {
    #pragma unroll
    for (int g = 0; g < 8; ++g) jbuf[wave * 16 + half * 8 + g] = bi[g];
  }
  __syncthreads();

  // gather x_q = emb[idx], write straight-through output, accumulate loss
  int row0 = blockIdx.x * ROWS_PER_WG;
  float lsum = 0.0f;
  for (int e = tid; e < ROWS_PER_WG * ED; e += blockDim.x) {
    int r = e >> 6, k = e & 63;
    int j = jbuf[r];
    float ev = emb[(size_t)j * ED + k];
    float xv = x[(size_t)(row0 + r) * ED + k];
    out_xq[(size_t)(row0 + r) * ED + k] = ev;   // x + sg(x_q - x) == x_q
    float d = ev - xv;
    lsum += d * d;
  }
  #pragma unroll
  for (int m = 1; m < 32; m <<= 1) lsum += __shfl_xor(lsum, m, 32);
  if (lane == 0) wsum[wave] = lsum;
  __syncthreads();
  if (tid == 0) {
    float t = 0.0f;
    for (int w = 0; w < 8; ++w) t += wsum[w];
    // loss = (1 + BETA) * mean((x_q - x)^2)
    atomicAdd(out_loss, t * (1.25f / (float)((size_t)TOK * ED)));
  }
  if (tid < ROWS_PER_WG) out_idx[row0 + tid] = (float)jbuf[tid];
}

// ------------------------------- launch -------------------------------------

extern "C" void kernel_launch(void* const* d_in, const int* in_sizes, int n_in,
                              void* d_out, int out_size, void* d_ws, size_t ws_size,
                              hipStream_t stream) {
  const float* x   = (const float*)d_in[0];
  const float* emb = (const float*)d_in[1];

  char* ws = (char*)d_ws;
  _Float16* xh  = (_Float16*)(ws + 0);            // 4,194,304 B
  _Float16* eh  = (_Float16*)(ws + 4194304);      //   262,144 B
  float* esq    = (float*)(ws + 4456448);         //     8,192 B
  float* alpha  = (float*)(ws + 4464640);         //   131,072 B
  float* tmin   = (float*)(ws + 4595712);         //   131,072 B
  float* u      = (float*)(ws + 4726784);         //   131,072 B
  float* c      = (float*)(ws + 4857856);         //     8,192 B
  float* lc     = (float*)(ws + 4866048);         //     8,192 B
  float* vpart  = (float*)(ws + 4874240);         // 2,097,152 B  (total ~6.97 MB)

  float* xq   = (float*)d_out;
  float* loss = (float*)d_out + (size_t)TOK * ED;
  float* idxf = loss + 1;

  prep_x_kernel<<<(TOK * ED) / 256, 256, 0, stream>>>(x, xh, u, loss);
  prep_emb_kernel<<<NE / 8, 256, 0, stream>>>(emb, eh, esq);
  rowstat_kernel<<<NSTRIPE, 256, SM_BYTES, stream>>>(xh, eh, esq, alpha, tmin);

  for (int it = 0; it < SKITERS; ++it) {
    colsum_kernel<<<NSTRIPE, 256, SM_BYTES, stream>>>(xh, eh, esq, alpha, tmin, u, vpart);
    reduce_c_kernel<<<NE / 256, 256, 0, stream>>>(vpart, c, lc);
    if (it + 1 < SKITERS)   // last row-normalization does not affect argmax
      rowsum_kernel<<<NSTRIPE, 256, SM_BYTES, stream>>>(xh, eh, esq, alpha, tmin, c, u);
  }

  argmax_out_kernel<<<NSTRIPE, 256, SM_BYTES, stream>>>(x, emb, xh, eh, esq, alpha,
                                                        tmin, lc, xq, loss, idxf);
}
